// MultiLayerGRU_24575802868346
// MI455X (gfx1250) — matrix operand order repaired and checked by reading, not compiled, not run on hardware
//
#include <hip/hip_runtime.h>
#include <hip/hip_bf16.h>
#include <math.h>

// ---------------------------------------------------------------------------
// Multi-layer GRU for MI455X (gfx1250), wave32, bf16 WMMA (16x16x32, f32 acc)
//   - batched input projection: full-device WMMA GEMM
//   - recurrence: ONE persistent kernel per layer; W_hh lives in LDS
//     (99 KB of the 320 KB/WGP); 4 waves/WG = (2 batch halves x 2 K halves)
//     so all 4 SIMD32s of the WGP work and the dependent WMMA chain per
//     timestep is halved; timesteps separated by a device-scope atomic
//     grid barrier instead of 512 kernel launches.
// ---------------------------------------------------------------------------

typedef __attribute__((ext_vector_type(16))) __bf16 v16bf;
typedef __attribute__((ext_vector_type(8)))  __bf16 v8bf;
typedef __attribute__((ext_vector_type(8)))  float  v8f;

static constexpr int GB = 32;     // batch
static constexpr int GT = 512;    // time steps
static constexpr int GH = 1024;   // hidden
static constexpr int GL = 4;      // layers
static constexpr int GG = 3 * GH; // 3H = 3072
static constexpr int NWG = GH / 16;        // 64 persistent workgroups
static constexpr int PADK = GH + 8;        // LDS row pitch (bank spread)
static constexpr int KHALF = GH / 2;       // 512

__device__ __forceinline__ float sigmoid_f(float x) {
    return 1.0f / (1.0f + __expf(-x));
}

// ---------------------------------------------------------------------------
// One-time fp32 -> bf16 weight conversion into workspace
// ---------------------------------------------------------------------------
__global__ void gru_convert_kernel(const float* __restrict__ wih,
                                   const float* __restrict__ whh,
                                   __bf16* __restrict__ wih_b,
                                   __bf16* __restrict__ whh_b,
                                   size_t n) {
    size_t i = (size_t)blockIdx.x * blockDim.x + threadIdx.x;
    size_t stride = (size_t)gridDim.x * blockDim.x;
    for (; i < n; i += stride) {
        wih_b[i] = (__bf16)wih[i];
        whh_b[i] = (__bf16)whh[i];
    }
}

// ---------------------------------------------------------------------------
// Zero h (fp32), h (bf16 copy), and the grid-barrier counter per layer
// ---------------------------------------------------------------------------
__global__ void gru_zeroh_kernel(float* __restrict__ h,
                                 __bf16* __restrict__ hbf, int n,
                                 unsigned int* __restrict__ bar) {
    int i = blockIdx.x * blockDim.x + threadIdx.x;
    if (i < n) {
        h[i] = 0.0f;
        hbf[i] = (__bf16)0.0f;
    }
    if (i == 0) *bar = 0u;
}

// ---------------------------------------------------------------------------
// Input projection: gi[M,3H] = X[M,H] @ Wih^T + b_ih   (M = B*T = 16384)
// Block: 256 threads = 8 waves; block tile 64(M) x 128(N);
// each wave: 16(M) x 64(N) via 4 accumulators, A-fragment reused 4x.
// ---------------------------------------------------------------------------
__global__ __launch_bounds__(256) void gru_proj_kernel(
    const float* __restrict__ X,       // [M, H] row-major (contiguous)
    const __bf16* __restrict__ Wih,    // [3H, H] bf16 row-major
    const float* __restrict__ b_ih,    // [3H]
    float* __restrict__ gi)            // [M, 3H]
{
    const int lane = threadIdx.x & 31;
    const int wave = threadIdx.x >> 5;       // 0..7
    const int mw = wave & 3;                 // 4 waves along M
    const int nw = wave >> 2;                // 2 waves along N
    const int m0 = blockIdx.x * 64 + mw * 16;
    const int n0 = blockIdx.y * 128 + nw * 64;

    // A-fragment addressing (16-bit A 16x32): lane half h -> 16 contiguous K at 16h
    const int ar = lane & 15;
    const int ak = (lane >> 4) * 16;
    // B-fragment addressing (16-bit B 32x16): lane = column, 16 contiguous K
    const int bc = lane & 15;
    const int bk = (lane >> 4) * 16;

    v8f acc[4] = {v8f{}, v8f{}, v8f{}, v8f{}};

    const float* xrow = X + (size_t)(m0 + ar) * GH;

#pragma unroll 4
    for (int k0 = 0; k0 < GH; k0 += 32) {
        // Load + convert A fragment (fp32 -> bf16)
        v16bf a;
#pragma unroll
        for (int i = 0; i < 16; ++i) {
            a[i]     = (__bf16)xrow[k0 + ak + i];
        }
#pragma unroll
        for (int nn = 0; nn < 4; ++nn) {
            const __bf16* wr = Wih + (size_t)(n0 + nn * 16 + bc) * GH + k0 + bk;
            v16bf b = *(const v16bf*)wr;   // 32B aligned: 2x global_load_b128
            acc[nn] = __builtin_amdgcn_wmma_f32_16x16x32_bf16(
                false, a, false, b, (short)0, acc[nn], false, false);
        }
    }

    // Store D + bias. D layout: VGPR v, lanes 0-15 -> M=v, lanes 16-31 -> M=v+8
    const int moff = (lane >> 4) * 8;
    const int dcol = lane & 15;
#pragma unroll
    for (int nn = 0; nn < 4; ++nn) {
        const int gn = n0 + nn * 16 + dcol;
        const float bias = b_ih[gn];
#pragma unroll
        for (int v = 0; v < 8; ++v) {
            const int gm = m0 + moff + v;
            gi[(size_t)gm * GG + gn] = acc[nn][v] + bias;
        }
    }
}

// ---------------------------------------------------------------------------
// Persistent recurrent-layer kernel.
// Grid: 64 WGs x 128 threads (4 waves). WG b = hidden columns [16b, 16b+16).
// Wave w: mhalf = w&1 (batch rows 0-15 / 16-31), khalf = w>>1 (K half).
// Each wave accumulates its K-half; khalf==1 waves spill partials to LDS,
// khalf==0 waves combine + run fused gate math and all global writes.
// ---------------------------------------------------------------------------
__global__ __launch_bounds__(128) void gru_layer_kernel(
    const float* __restrict__ gi,        // [B*T, 3H] (row = b*T + t)
    const __bf16* __restrict__ Whh,      // [3H, H] bf16
    const float* __restrict__ b_hh,      // [3H]
    float* __restrict__ h,               // [B, H] fp32 state
    __bf16* __restrict__ hbf0,           // [B, H] bf16 ping (zeroed = h0)
    __bf16* __restrict__ hbf1,           // [B, H] bf16 pong
    float* __restrict__ y,               // layer output [B, T, H]
    float* __restrict__ hx,              // [B, H] final hidden
    unsigned int* __restrict__ bar)      // grid barrier counter (zeroed)
{
    __shared__ __bf16 ldsW[3 * 16 * PADK];        // 99,072 B weight slice
    __shared__ float  ldsP[2][3][8][32];          // 6,144 B K-partials

    const int lane = threadIdx.x & 31;
    const int wave = threadIdx.x >> 5;  // 0..3
    const int mhalf = wave & 1;         // batch half
    const int khalf = wave >> 1;        // K half
    const int m0 = mhalf * 16;          // batch row base
    const int n0 = blockIdx.x * 16;     // hidden column base (this WG)

    // ---- Preload W_hh slice into LDS (rows: gate*16 + rr, pitch PADK) ----
    for (int idx = threadIdx.x; idx < 48 * (GH / 8); idx += 128) {
        const int row = idx / (GH / 8);          // 0..47
        const int kc  = (idx % (GH / 8)) * 8;    // K chunk of 8
        const int g   = row >> 4;
        const int rr  = row & 15;
        v8bf v = *(const v8bf*)(Whh + ((size_t)(g * GH + n0 + rr)) * GH + kc);
        *(v8bf*)&ldsW[row * PADK + kc] = v;
    }
    __syncthreads();

    // Fragment addressing
    const int ar = lane & 15;
    const int ak = (lane >> 4) * 16;
    const int bc = lane & 15;
    const int bk = (lane >> 4) * 16;
    const int moff = (lane >> 4) * 8;
    const int j = n0 + (lane & 15);

    const float bhr = b_hh[j];
    const float bhz = b_hh[GH + j];
    const float bhn = b_hh[2 * GH + j];

    const __bf16* lR = &ldsW[(0 * 16 + bc) * PADK];
    const __bf16* lZ = &ldsW[(1 * 16 + bc) * PADK];
    const __bf16* lN = &ldsW[(2 * 16 + bc) * PADK];
    const int kbeg = khalf * KHALF;

    for (int t = 0; t < GT; ++t) {
        const __bf16* hin = (t & 1) ? hbf1 : hbf0;
        __bf16* hout      = (t & 1) ? hbf0 : hbf1;

        v8f accR = {}, accZ = {}, accN = {};
        const __bf16* hrow = hin + (size_t)(m0 + ar) * GH;

#pragma unroll 4
        for (int kk = 0; kk < KHALF; kk += 32) {
            const int k0 = kbeg + kk;
            v16bf a;
            v8bf alo = *(const v8bf*)(hrow + k0 + ak);
            v8bf ahi = *(const v8bf*)(hrow + k0 + ak + 8);
#pragma unroll
            for (int i = 0; i < 8; ++i) { a[i] = alo[i]; a[8 + i] = ahi[i]; }

            // B fragments from LDS (16B-aligned v8bf pairs -> ds_load_b128)
            v16bf br, bz, bn;
            v8bf r0 = *(const v8bf*)(lR + k0 + bk);
            v8bf r1 = *(const v8bf*)(lR + k0 + bk + 8);
            v8bf z0 = *(const v8bf*)(lZ + k0 + bk);
            v8bf z1 = *(const v8bf*)(lZ + k0 + bk + 8);
            v8bf n0v = *(const v8bf*)(lN + k0 + bk);
            v8bf n1v = *(const v8bf*)(lN + k0 + bk + 8);
#pragma unroll
            for (int i = 0; i < 8; ++i) {
                br[i] = r0[i]; br[8 + i] = r1[i];
                bz[i] = z0[i]; bz[8 + i] = z1[i];
                bn[i] = n0v[i]; bn[8 + i] = n1v[i];
            }
            accR = __builtin_amdgcn_wmma_f32_16x16x32_bf16(
                false, a, false, br, (short)0, accR, false, false);
            accZ = __builtin_amdgcn_wmma_f32_16x16x32_bf16(
                false, a, false, bz, (short)0, accZ, false, false);
            accN = __builtin_amdgcn_wmma_f32_16x16x32_bf16(
                false, a, false, bn, (short)0, accN, false, false);
        }

        // ---- Combine K-halves: khalf==1 spills, khalf==0 reduces ----
        if (khalf == 1) {
#pragma unroll
            for (int v = 0; v < 8; ++v) {
                ldsP[mhalf][0][v][lane] = accR[v];
                ldsP[mhalf][1][v][lane] = accZ[v];
                ldsP[mhalf][2][v][lane] = accN[v];
            }
        }
        __syncthreads();

        if (khalf == 0) {
            // Fused gate math + state update (columns owned by this WG)
#pragma unroll
            for (int v = 0; v < 8; ++v) {
                const float aR = accR[v] + ldsP[mhalf][0][v][lane];
                const float aZ = accZ[v] + ldsP[mhalf][1][v][lane];
                const float aN = accN[v] + ldsP[mhalf][2][v][lane];
                const int b = m0 + moff + v;                 // batch index
                const size_t girow = ((size_t)b * GT + t) * GG;
                const float r = sigmoid_f(gi[girow + j]         + aR + bhr);
                const float z = sigmoid_f(gi[girow + GH + j]    + aZ + bhz);
                const float n = tanhf(   gi[girow + 2 * GH + j] + r * (aN + bhn));
                const size_t hidx = (size_t)b * GH + j;
                const float hp = h[hidx];
                const float hnew = (1.0f - z) * n + z * hp;
                h[hidx] = hnew;
                hout[hidx] = (__bf16)hnew;
                y[((size_t)b * GT + t) * GH + j] = hnew;
                if (t == GT - 1) hx[hidx] = hnew;
            }
        }

        // ---- Device-scope grid barrier between timesteps ----
        // (its __syncthreads also protects ldsP against next-step WAR)
        if (t != GT - 1) {
            __threadfence();                 // release: h/hout visible at L2
            __syncthreads();
            if (threadIdx.x == 0) {
                const unsigned int target = (unsigned int)(t + 1) * NWG;
                __hip_atomic_fetch_add(bar, 1u, __ATOMIC_RELEASE,
                                       __HIP_MEMORY_SCOPE_AGENT);
                while (__hip_atomic_load(bar, __ATOMIC_ACQUIRE,
                                         __HIP_MEMORY_SCOPE_AGENT) < target) {
                    __builtin_amdgcn_s_sleep(1);
                }
            }
            __syncthreads();
            __threadfence();                 // acquire: invalidate stale L0
        }
    }
}

// ---------------------------------------------------------------------------
// Host launcher
// ---------------------------------------------------------------------------
extern "C" void kernel_launch(void* const* d_in, const int* in_sizes, int n_in,
                              void* d_out, int out_size, void* d_ws, size_t ws_size,
                              hipStream_t stream) {
    const float* X0  = (const float*)d_in[0];   // [B, T, H]
    const float* Wih = (const float*)d_in[1];   // [L, 3H, H]
    const float* Whh = (const float*)d_in[2];   // [L, 3H, H]
    const float* bih = (const float*)d_in[3];   // [L, 3H]
    const float* bhh = (const float*)d_in[4];   // [L, 3H]
    float* out = (float*)d_out;                 // [L,B,T,H] ++ [L,B,H]

    // Workspace carve-up (all 32B-aligned by construction)
    const size_t wsz = (size_t)GL * GG * GH;        // elems per weight tensor
    __bf16* WihB = (__bf16*)d_ws;                   // 25.2 MB
    __bf16* WhhB = WihB + wsz;                      // 25.2 MB
    float*  gi   = (float*)(WhhB + wsz);            // B*T*3H fp32 = 201 MB
    float*  h    = gi + (size_t)GB * GT * GG;       // 128 KB
    __bf16* hbf0 = (__bf16*)(h + GB * GH);          // 64 KB
    __bf16* hbf1 = hbf0 + GB * GH;                  // 64 KB
    unsigned int* bar = (unsigned int*)(hbf1 + GB * GH);

    // 1) Convert weights to bf16 once per launch (deterministic)
    gru_convert_kernel<<<2048, 256, 0, stream>>>(Wih, Whh, WihB, WhhB, wsz);

    const size_t layer_out_elems = (size_t)GB * GT * GH;
    float* hx_base = out + (size_t)GL * layer_out_elems;

    for (int l = 0; l < GL; ++l) {
        const float* X = (l == 0) ? X0 : (out + (size_t)(l - 1) * layer_out_elems);
        float* y  = out + (size_t)l * layer_out_elems;
        float* hx = hx_base + (size_t)l * GB * GH;

        // 2) Batched input projection: gi = X @ Wih[l]^T + b_ih[l]
        gru_proj_kernel<<<dim3(GB * GT / 64, GG / 128), 256, 0, stream>>>(
            X, WihB + (size_t)l * GG * GH, bih + (size_t)l * GG, gi);

        // 3) h0 = 0, barrier counter = 0
        gru_zeroh_kernel<<<(GB * GH + 255) / 256, 256, 0, stream>>>(
            h, hbf0, GB * GH, bar);

        // 4) Entire recurrent scan in ONE persistent kernel
        gru_layer_kernel<<<NWG, 128, 0, stream>>>(
            gi, WhhB + (size_t)l * GG * GH, bhh + (size_t)l * GG,
            h, hbf0, hbf1, y, hx, bar);
    }
}
